// Model_mHC_85298050498983
// MI455X (gfx1250) — matrix operand order repaired and checked by reading, not compiled, not run on hardware
//
#include <hip/hip_runtime.h>
#include <hip/hip_bf16.h>
#include <math.h>

typedef __attribute__((ext_vector_type(16))) __bf16 v16bf;
typedef __attribute__((ext_vector_type(8)))  __bf16 v8bf;
typedef __attribute__((ext_vector_type(8)))  float  v8f;

#define N_STREAMS 4
#define N_HEADS   4
#define D_STATE   64
#define EPS_RMS   1.1920929e-07f

// ---------------------------------------------------------------------------
// f32 [K][N] -> bf16 [N][K] transposed conversion, LDS-tiled 32x32.
// blockDim (32,8); grid (ceil(N/32), K/32, layers)
// ---------------------------------------------------------------------------
__global__ __launch_bounds__(256)
void k_cvt_transpose(const float* __restrict__ in, __bf16* __restrict__ out,
                     int K, int N) {
  __shared__ __bf16 tile[32][33];
  const size_t z = blockIdx.z;
  in  += z * (size_t)K * N;
  out += z * (size_t)N * K;
  const int tx = threadIdx.x, ty = threadIdx.y;
  const int k0 = blockIdx.y * 32, n0 = blockIdx.x * 32;
#pragma unroll
  for (int i = 0; i < 4; ++i) {
    int k = ty + i * 8;
    int gn = n0 + tx;
    float v = (gn < N) ? in[(size_t)(k0 + k) * N + gn] : 0.f;
    tile[k][tx] = (__bf16)v;
  }
  __syncthreads();
#pragma unroll
  for (int i = 0; i < 4; ++i) {
    int n = n0 + ty + i * 8;
    if (n < N) out[(size_t)n * K + (k0 + tx)] = tile[tx][ty + i * 8];
  }
}

// ---------------------------------------------------------------------------
// embedding gather: H[row, :] = embed[tokens[row], :]   (d = 1024 -> float4)
// ---------------------------------------------------------------------------
__global__ __launch_bounds__(256)
void k_embed(const int* __restrict__ tokens, const float* __restrict__ embed,
             float* __restrict__ H) {
  int row = blockIdx.x;
  int tok = tokens[row];
  const float4* src = (const float4*)(embed + (size_t)tok * 1024);
  float4* dst = (float4*)(H + (size_t)row * 1024);
  dst[threadIdx.x] = src[threadIdx.x];
}

// ---------------------------------------------------------------------------
// RMSNorm per row
// ---------------------------------------------------------------------------
__global__ __launch_bounds__(256) void k_rmsnorm(const float* __restrict__ X,
                                                 const float* __restrict__ w,
                                                 float* __restrict__ Y, int d) {
  __shared__ float buf[256];
  int row = blockIdx.x;
  const float* x = X + (size_t)row * d;
  float* y = Y + (size_t)row * d;
  float ss = 0.f;
  for (int c = threadIdx.x; c < d; c += 256) { float v = x[c]; ss += v * v; }
  buf[threadIdx.x] = ss;
  __syncthreads();
  for (int s = 128; s > 0; s >>= 1) {
    if (threadIdx.x < s) buf[threadIdx.x] += buf[threadIdx.x + s];
    __syncthreads();
  }
  float scale = rsqrtf(buf[0] / (float)d + EPS_RMS);
  for (int c = threadIdx.x; c < d; c += 256) y[c] = x[c] * scale * w[c];
}

// ---------------------------------------------------------------------------
// Sinkhorn mixing matrix (4x4, 5 iterations) -> Mout[16]
// ---------------------------------------------------------------------------
__global__ void k_sinkhorn(const float* __restrict__ logits, float* __restrict__ Mout) {
  if (threadIdx.x != 0 || blockIdx.x != 0) return;
  float M[16];
  for (int i = 0; i < 16; ++i) {
    int r = i >> 2, c = i & 3;
    M[i] = 1.f / (1.f + expf(-logits[i])) + (r == c ? 1.f : 0.f);
  }
  for (int it = 0; it < 5; ++it) {
    for (int r = 0; r < 4; ++r) {
      float s = M[r*4+0] + M[r*4+1] + M[r*4+2] + M[r*4+3] + 1e-6f;
      for (int c = 0; c < 4; ++c) M[r*4+c] /= s;
    }
    for (int c = 0; c < 4; ++c) {
      float s = M[0*4+c] + M[1*4+c] + M[2*4+c] + M[3*4+c] + 1e-6f;
      for (int r = 0; r < 4; ++r) M[r*4+c] /= s;
    }
  }
  for (int i = 0; i < 16; ++i) Mout[i] = M[i];
}

// ---------------------------------------------------------------------------
// bf16 WMMA GEMM: C[M,N] = A[M,K](f32) @ W[K,N] + bias[N], W given transposed
// bf16 as Wt[N][K].  Block tile 128x64, BK=32, 256 threads = 8 waves
// (4 M-waves x 2 N-waves), each wave computes a 32x32 tile via 4
// v_wmma_f32_16x16x32_bf16 accumulators.  Double-buffered LDS with register
// prefetch: global loads for tile kt+1 issue before the WMMAs of tile kt.
// M, K multiples of 128/32 at all call sites; N guarded.
// ---------------------------------------------------------------------------
#define GBM 128
#define GBN 64
#define GBK 32
#define LSTR 40  // padded LDS row stride in bf16 elements (80B, 16B-aligned)

static __device__ __forceinline__ v16bf cat16(v8bf lo, v8bf hi) {
  return __builtin_shufflevector(lo, hi, 0,1,2,3,4,5,6,7,8,9,10,11,12,13,14,15);
}

__global__ __launch_bounds__(256)
void k_gemm_bf16(const float* __restrict__ A, const __bf16* __restrict__ Wt,
                 const float* __restrict__ bias, float* __restrict__ C,
                 int M, int N, int K) {
  __shared__ __align__(16) __bf16 As[2][GBM * LSTR];
  __shared__ __align__(16) __bf16 Bs[2][GBN * LSTR];
  const int tid  = threadIdx.x;
  const int m0   = blockIdx.y * GBM;
  const int n0   = blockIdx.x * GBN;
  const int lane = tid & 31;
  const int wv   = tid >> 5;
  const int wm   = wv & 3;   // M-wave 0..3 -> rows wm*32..+32
  const int wn   = wv >> 2;  // N-wave 0..1 -> cols wn*32..+32
  const int half = lane >> 4;
  const int l15  = lane & 15;

  // staging coordinates: A rows ar, ar+64 (8 cols each); B row bn (8 ks)
  const int ar = tid >> 2;
  const int ac = (tid & 3) * 8;
  const int bn = tid >> 2;
  const int bk = (tid & 3) * 8;
  const bool bvalid = (n0 + bn) < N;

  float4 a00, a01, a10, a11;
  v8bf pb;

  // ---- prefetch tile k0 into registers ----
  auto prefetch = [&](int k0) {
    const float* ap0 = &A[(size_t)(m0 + ar) * K + k0 + ac];
    const float* ap1 = &A[(size_t)(m0 + ar + 64) * K + k0 + ac];
    a00 = *(const float4*)(ap0);
    a01 = *(const float4*)(ap0 + 4);
    a10 = *(const float4*)(ap1);
    a11 = *(const float4*)(ap1 + 4);
    if (bvalid) pb = *(const v8bf*)&Wt[(size_t)(n0 + bn) * K + k0 + bk];
    else        pb = (v8bf)(__bf16)0.0f;
  };
  // ---- convert + commit registers into LDS buffer `buf` ----
  auto commit = [&](int buf) {
    v8bf c0, c1;
    c0[0]=(__bf16)a00.x; c0[1]=(__bf16)a00.y; c0[2]=(__bf16)a00.z; c0[3]=(__bf16)a00.w;
    c0[4]=(__bf16)a01.x; c0[5]=(__bf16)a01.y; c0[6]=(__bf16)a01.z; c0[7]=(__bf16)a01.w;
    c1[0]=(__bf16)a10.x; c1[1]=(__bf16)a10.y; c1[2]=(__bf16)a10.z; c1[3]=(__bf16)a10.w;
    c1[4]=(__bf16)a11.x; c1[5]=(__bf16)a11.y; c1[6]=(__bf16)a11.z; c1[7]=(__bf16)a11.w;
    *(v8bf*)&As[buf][ar * LSTR + ac]        = c0;
    *(v8bf*)&As[buf][(ar + 64) * LSTR + ac] = c1;
    *(v8bf*)&Bs[buf][bn * LSTR + bk]        = pb;
  };

  v8f acc00 = {}, acc01 = {}, acc10 = {}, acc11 = {};

  prefetch(0);
  commit(0);
  __syncthreads();

  const int nK = K / GBK;
  for (int kt = 0; kt < nK; ++kt) {
    const bool more = (kt + 1) < nK;
    if (more) prefetch((kt + 1) * GBK);  // global loads in flight over WMMAs

    const int cur = kt & 1;
    // A fragments: rows wm*32 + mi*16 + l15; K chunks at half*8 and 16+half*8
    const __bf16* ap0 = &As[cur][(wm * 32 + l15) * LSTR + half * 8];
    const __bf16* ap1 = ap0 + 16 * LSTR;
    v16bf af0 = cat16(*(const v8bf*)ap0, *(const v8bf*)(ap0 + 16));
    v16bf af1 = cat16(*(const v8bf*)ap1, *(const v8bf*)(ap1 + 16));
    // B fragments: rows wn*32 + ni*16 + l15; K chunks at half*8 and 16+half*8
    const __bf16* bp0 = &Bs[cur][(wn * 32 + l15) * LSTR + half * 8];
    const __bf16* bp1 = bp0 + 16 * LSTR;
    v16bf bf0 = cat16(*(const v8bf*)bp0, *(const v8bf*)(bp0 + 16));
    v16bf bf1 = cat16(*(const v8bf*)bp1, *(const v8bf*)(bp1 + 16));

    acc00 = __builtin_amdgcn_wmma_f32_16x16x32_bf16(false, af0, false, bf0,
                                                    (short)0, acc00, false, false);
    acc01 = __builtin_amdgcn_wmma_f32_16x16x32_bf16(false, af0, false, bf1,
                                                    (short)0, acc01, false, false);
    acc10 = __builtin_amdgcn_wmma_f32_16x16x32_bf16(false, af1, false, bf0,
                                                    (short)0, acc10, false, false);
    acc11 = __builtin_amdgcn_wmma_f32_16x16x32_bf16(false, af1, false, bf1,
                                                    (short)0, acc11, false, false);

    if (more) commit((kt + 1) & 1);
    __syncthreads();
  }

  // C/D layout: n = l15 (+tile base), m = half*8 + v
  const int nA = n0 + wn * 32 + l15;
  const int nB = nA + 16;
  const int mA = m0 + wm * 32 + half * 8;
  const int mB = mA + 16;
#pragma unroll
  for (int v = 0; v < 8; ++v) {
    if (nA < N) {
      C[(size_t)(mA + v) * N + nA] = acc00[v] + bias[nA];
      C[(size_t)(mB + v) * N + nA] = acc10[v] + bias[nA];
    }
    if (nB < N) {
      C[(size_t)(mA + v) * N + nB] = acc01[v] + bias[nB];
      C[(size_t)(mB + v) * N + nB] = acc11[v] + bias[nB];
    }
  }
}

// ---------------------------------------------------------------------------
// scan prep: dt = softplus(dt_raw + dt_bias); dtA = exp(dt * -exp(A_log));
// dtB[n] = dt * B_mat[n].  One block per (b,l) row; 256 threads = 4 heads x 64.
// zxbcdt row: [0:2048)=z [2048:4096)=x [4096:4100)=dt [4100:4612)=BC
// ---------------------------------------------------------------------------
__global__ __launch_bounds__(256)
void k_prep(const float* __restrict__ zx, const float* __restrict__ A_log,
            const float* __restrict__ dt_bias, float* __restrict__ dtA,
            float* __restrict__ dtB) {
  int row = blockIdx.x;
  int h = threadIdx.x >> 6;
  int n = threadIdx.x & 63;
  const float* r = zx + (size_t)row * 4612;
  float dtr = r[4096 + h] + dt_bias[h];
  float dts = fmaxf(dtr, 0.f) + log1pf(expf(-fabsf(dtr)));  // stable softplus
  float Ah  = -expf(A_log[h]);
  if (n == 0) dtA[row * N_HEADS + h] = expf(dts * Ah);
  dtB[((size_t)row * N_HEADS + h) * D_STATE + n] = dts * r[4100 + h * 128 + n];
}

// ---------------------------------------------------------------------------
// selective scan: each thread owns one p-column, h_state[64] in registers.
// grid: (P/256, N_HEADS, B), 256 threads.
// ---------------------------------------------------------------------------
__global__ __launch_bounds__(256)
void k_scan(const float* __restrict__ zx, const float* __restrict__ dtA,
            const float* __restrict__ dtB, float* __restrict__ ybuf, int L) {
  __shared__ float sdtB[D_STATE];
  __shared__ float sC[D_STATE];
  __shared__ float sdtA;
  const int tid = threadIdx.x;
  const int hh  = blockIdx.y;
  const int b   = blockIdx.z;
  const int p   = blockIdx.x * 256 + tid;

  float h_state[D_STATE];
#pragma unroll
  for (int n = 0; n < D_STATE; ++n) h_state[n] = 0.f;

  for (int t = 0; t < L; ++t) {
    size_t row = (size_t)(b * L + t);
    const float* r = zx + row * 4612;
    if (tid < 64) sdtB[tid] = dtB[(row * N_HEADS + hh) * D_STATE + tid];
    else if (tid < 128) sC[tid - 64] = r[4100 + hh * 128 + 64 + (tid - 64)];
    else if (tid == 128) sdtA = dtA[row * N_HEADS + hh];
    __syncthreads();

    float xv = r[2048 + hh * 512 + p];
    float da = sdtA;
    float y = 0.f;
#pragma unroll
    for (int n = 0; n < D_STATE; ++n) {
      h_state[n] = fmaf(h_state[n], da, sdtB[n] * xv);
      y = fmaf(sC[n], h_state[n], y);
    }
    ybuf[row * 2048 + hh * 512 + p] = y;
    __syncthreads();
  }
}

// ---------------------------------------------------------------------------
// gate: y = (y + x*D[h]) * silu(z), in place on ybuf
// ---------------------------------------------------------------------------
__global__ void k_gate(const float* __restrict__ zx, float* __restrict__ ybuf,
                       const float* __restrict__ Dl, long total) {
  long i = (long)blockIdx.x * blockDim.x + threadIdx.x;
  long stride = (long)gridDim.x * blockDim.x;
  for (; i < total; i += stride) {
    long row = i >> 11;          // /2048
    int  c   = (int)(i & 2047);
    int  h   = c >> 9;
    const float* r = zx + (size_t)row * 4612;
    float z = r[c];
    float x = r[2048 + c];
    float y = ybuf[i] + x * Dl[h];
    float sil = z / (1.f + expf(-z));
    ybuf[i] = y * sil;
  }
}

// ---------------------------------------------------------------------------
// combine: H[row, j*256+d] = xout[row, j*256+d] + sum_i M[j,i]*H[row, i*256+d]
// ---------------------------------------------------------------------------
__global__ __launch_bounds__(256)
void k_combine(float* __restrict__ H, const float* __restrict__ xout,
               const float* __restrict__ Mws) {
  __shared__ float srow[1024];
  __shared__ float sM[16];
  int row = blockIdx.x;
  if (threadIdx.x < 16) sM[threadIdx.x] = Mws[threadIdx.x];
  for (int c = threadIdx.x; c < 1024; c += 256) srow[c] = H[(size_t)row * 1024 + c];
  __syncthreads();
  for (int c = threadIdx.x; c < 1024; c += 256) {
    int j = c >> 8, d = c & 255;
    float acc = xout[(size_t)row * 1024 + c];
#pragma unroll
    for (int i = 0; i < 4; ++i) acc = fmaf(sM[j * 4 + i], srow[i * 256 + d], acc);
    H[(size_t)row * 1024 + c] = acc;
  }
}

// ---------------------------------------------------------------------------
// launch
// ---------------------------------------------------------------------------
extern "C" void kernel_launch(void* const* d_in, const int* in_sizes, int n_in,
                              void* d_out, int out_size, void* d_ws, size_t ws_size,
                              hipStream_t stream) {
  (void)in_sizes; (void)n_in; (void)out_size; (void)ws_size;

  const int Bb = 2, Ll = 2048, Dm = 1024, NL = 4;
  const int Dproj = 4612, Din = 2048, Vocab = 32000;
  const int Rows = Bb * Ll;  // 4096

  const int*   tokens     = (const int*)  d_in[0];
  const float* embed      = (const float*)d_in[1];
  const float* norm_w     = (const float*)d_in[2];
  const float* in_proj_W  = (const float*)d_in[3];
  const float* in_proj_b  = (const float*)d_in[4];
  const float* A_log      = (const float*)d_in[5];
  const float* Dparam     = (const float*)d_in[6];
  const float* dt_bias    = (const float*)d_in[7];
  const float* out_proj_W = (const float*)d_in[8];
  const float* out_proj_b = (const float*)d_in[9];
  const float* mix_logits = (const float*)d_in[10];
  const float* norm_f_w   = (const float*)d_in[11];
  const float* head_W     = (const float*)d_in[12];
  const float* head_b     = (const float*)d_in[13];
  float* out = (float*)d_out;

  // workspace layout (256B aligned slices)
  char* base = (char*)d_ws;
  size_t off = 0;
  auto take = [&](size_t bytes) -> char* {
    char* p = base + off;
    off = (off + bytes + 255) & ~(size_t)255;
    return p;
  };
  __bf16* bf_in   = (__bf16*)take((size_t)NL * Dm * Dproj * 2);   // [l][N=4612][K=1024]
  __bf16* bf_out  = (__bf16*)take((size_t)NL * Din * Dm * 2);     // [l][N=1024][K=2048]
  __bf16* bf_head = (__bf16*)take((size_t)Dm * Vocab * 2);        // [N=32000][K=1024]
  float*  Hbuf    = (float*) take((size_t)Rows * Dm * 4);
  float*  un      = (float*) take((size_t)Rows * Dm * 4);
  float*  zx      = (float*) take((size_t)Rows * Dproj * 4);
  float*  dtA     = (float*) take((size_t)Rows * N_HEADS * 4);
  float*  dtB     = (float*) take((size_t)Rows * N_HEADS * D_STATE * 4);
  float*  ybuf    = (float*) take((size_t)Rows * Din * 4);
  float*  xout    = (float*) take((size_t)Rows * Dm * 4);
  float*  Mws     = (float*) take(16 * 4);

  // weight conversion f32 [K][N] -> bf16 [N][K] (transposed, K-contiguous)
  dim3 tb(32, 8);
  k_cvt_transpose<<<dim3((Dproj + 31) / 32, Dm / 32, NL), tb, 0, stream>>>(
      in_proj_W, bf_in, Dm, Dproj);
  k_cvt_transpose<<<dim3(Dm / 32, Din / 32, NL), tb, 0, stream>>>(
      out_proj_W, bf_out, Din, Dm);
  k_cvt_transpose<<<dim3(Vocab / 32, Dm / 32, 1), tb, 0, stream>>>(
      head_W, bf_head, Dm, Vocab);

  // embedding
  k_embed<<<Rows, 256, 0, stream>>>(tokens, embed, Hbuf);

  for (int l = 0; l < NL; ++l) {
    k_sinkhorn<<<1, 32, 0, stream>>>(mix_logits + (size_t)l * 16, Mws);
    k_rmsnorm<<<Rows, 256, 0, stream>>>(Hbuf, norm_w + (size_t)l * Dm, un, Dm);

    // in_proj GEMM: (4096 x 1024) @ (1024 x 4612)
    k_gemm_bf16<<<dim3((Dproj + GBN - 1) / GBN, Rows / GBM), 256, 0, stream>>>(
        un, bf_in + (size_t)l * Dm * Dproj, in_proj_b + (size_t)l * Dproj, zx,
        Rows, Dproj, Dm);

    k_prep<<<Rows, 256, 0, stream>>>(zx, A_log + (size_t)l * N_HEADS,
                                     dt_bias + (size_t)l * N_HEADS, dtA, dtB);

    k_scan<<<dim3(2, N_HEADS, Bb), 256, 0, stream>>>(zx, dtA, dtB, ybuf, Ll);

    k_gate<<<4096, 256, 0, stream>>>(zx, ybuf, Dparam + (size_t)l * N_HEADS,
                                     (long)Rows * Din);

    // out_proj GEMM: (4096 x 2048) @ (2048 x 1024)
    k_gemm_bf16<<<dim3(Dm / GBN, Rows / GBM), 256, 0, stream>>>(
        ybuf, bf_out + (size_t)l * Din * Dm, out_proj_b + (size_t)l * Dm, xout,
        Rows, Dm, Din);

    k_combine<<<Rows, 256, 0, stream>>>(Hbuf, xout, Mws);
  }

  // final norm + LM head: (4096 x 1024) @ (1024 x 32000) -> d_out
  k_rmsnorm<<<Rows, 256, 0, stream>>>(Hbuf, norm_f_w, un, Dm);
  k_gemm_bf16<<<dim3(Vocab / GBN, Rows / GBM), 256, 0, stream>>>(
      un, bf_head, head_b, out, Rows, Vocab, Dm);
}
